// OCRLayer_52037823758594
// MI455X (gfx1250) — matrix operand order repaired and checked by reading, not compiled, not run on hardware
//
#include <hip/hip_runtime.h>

#define N_PTS   100000
#define N_PER   25000
#define NB      4
#define C       128
#define KK      32
#define HEADS   8
#define GRID3   128
#define MT      64                      // rows per WMMA block tile
#define TILES   ((N_PTS + MT - 1) / MT) // 1563
#define N_PAD   (TILES * MT)            // 100032 (padded row count)

typedef __attribute__((ext_vector_type(8)))  unsigned short us8;
typedef __attribute__((ext_vector_type(16))) __bf16         v16bf;
typedef __attribute__((ext_vector_type(8)))  float          v8f;
typedef __attribute__((ext_vector_type(4)))  int            v4i;

union BF16x16 { v16bf v; us8 h[2]; unsigned short u[16]; };

#if __has_builtin(__builtin_amdgcn_global_load_async_to_lds_b128)
#define HAVE_ASYNC_LDS 1
#else
#define HAVE_ASYNC_LDS 0
#endif

static __device__ __forceinline__ void wait_asynccnt0() {
#if HAVE_ASYNC_LDS
#if __has_builtin(__builtin_amdgcn_s_wait_asynccnt)
    __builtin_amdgcn_s_wait_asynccnt(0);
#else
    asm volatile("s_wait_asynccnt 0x0" ::: "memory");
#endif
#endif
}

static __device__ __forceinline__ void async_copy16(const unsigned short* src,
                                                    unsigned short* dst_lds) {
#if HAVE_ASYNC_LDS
    __builtin_amdgcn_global_load_async_to_lds_b128(
        (__attribute__((address_space(1))) v4i*)src,
        (__attribute__((address_space(3))) v4i*)dst_lds, 0, 0);
#else
    *(us8*)dst_lds = *(const us8*)src;
#endif
}

static __device__ __forceinline__ unsigned short f2bf(float f) {
    unsigned u = __float_as_uint(f);
    unsigned r = u + 0x7FFFu + ((u >> 16) & 1u);
    return (unsigned short)(r >> 16);
}

// ---------------------------------------------------------------- utilities
__global__ void fill_i32(int* p, int v, long n) {
    long i = (long)blockIdx.x * blockDim.x + threadIdx.x;
    long stride = (long)gridDim.x * blockDim.x;
    for (; i < n; i += stride) p[i] = v;
}
__global__ void fill_f32(float* p, float v, long n) {
    long i = (long)blockIdx.x * blockDim.x + threadIdx.x;
    long stride = (long)gridDim.x * blockDim.x;
    for (; i < n; i += stride) p[i] = v;
}

// ---------------------------------------------------------------- grid/nbr
__global__ void scatter_grid(const int* __restrict__ coords, int* __restrict__ grid) {
    int n = blockIdx.x * 256 + threadIdx.x;
    if (n >= N_PTS) return;
    const int* c4 = coords + (size_t)n * 4;
    size_t idx = (((size_t)c4[0] * GRID3 + c4[1]) * GRID3 + c4[2]) * GRID3 + c4[3];
    grid[idx] = n;
}

// nbr is padded to N_PAD rows: rows >= N_PTS get the zero-row sentinel, so the
// conv loader needs no bounds check in its hot loop.
__global__ void build_nbr(const int* __restrict__ coords, const int* __restrict__ grid,
                          int* __restrict__ nbr) {
    long i = (long)blockIdx.x * 256 + threadIdx.x;
    if (i >= (long)N_PAD * 27) return;
    int n = (int)(i / 27), tap = (int)(i % 27);
    int j = N_PTS;  // sentinel zero row
    if (n < N_PTS) {
        int dx = tap / 9 - 1, dy = (tap / 3) % 3 - 1, dz = tap % 3 - 1;
        const int* c4 = coords + (size_t)n * 4;
        int b = c4[0], x = c4[1] + dx, y = c4[2] + dy, z = c4[3] + dz;
        if ((unsigned)x < (unsigned)GRID3 && (unsigned)y < (unsigned)GRID3 &&
            (unsigned)z < (unsigned)GRID3) {
            int g = grid[(((size_t)b * GRID3 + x) * GRID3 + y) * GRID3 + z];
            if (g >= 0) j = g;
        }
    }
    nbr[i] = j;
}

// ---------------------------------------------------------------- conversions
__global__ void feats_to_bf16(const float* __restrict__ feats, unsigned short* __restrict__ fb) {
    long i = (long)blockIdx.x * 256 + threadIdx.x;
    if (i >= (long)(N_PTS + 1) * C) return;
    long n = i >> 7;
    float v = (n < N_PTS) ? feats[i] : 0.0f;
    fb[i] = f2bf(v);
}

// conv_w [27][cin][cout] -> wT [27][cout][cin] bf16
__global__ void prep_convw(const float* __restrict__ cw, unsigned short* __restrict__ wT) {
    long i = (long)blockIdx.x * 256 + threadIdx.x;
    if (i >= 27L * C * C) return;
    int tap = (int)(i / (C * C));
    int r   = (int)(i % (C * C));
    int co = r >> 7, ci = r & 127;
    wT[((size_t)tap * C + co) * C + ci] = f2bf(cw[((size_t)tap * C + ci) * C + co]);
}

// wq_b (in_proj rows 0..127), wout_b, wbeff (bottleneck halves folded)
__global__ void prep_small(const float* __restrict__ inw, const float* __restrict__ outw,
                           const float* __restrict__ bw,
                           unsigned short* __restrict__ wq, unsigned short* __restrict__ wo,
                           unsigned short* __restrict__ wb) {
    int i = blockIdx.x * 256 + threadIdx.x;
    if (i >= 3 * C * C) return;
    int sel = i / (C * C), r = i % (C * C);
    int co = r >> 7, ci = r & 127;
    if (sel == 0)      wq[r] = f2bf(inw[(size_t)co * C + ci]);
    else if (sel == 1) wo[r] = f2bf(outw[(size_t)co * C + ci]);
    else               wb[r] = f2bf(bw[(size_t)co * 2 * C + ci] + bw[(size_t)co * 2 * C + C + ci]);
}

// ---------------------------------------------------------------- conv WMMA
// 64x128 tile/block, 8 waves, 4 accumulators/wave.  Double-buffered LDS with
// async global->LDS gather of tap t+1 overlapped with the 16 WMMAs of tap t.
// One s_wait_asynccnt + one barrier per tap; loader is branch-free (padded nbr).
__global__ __launch_bounds__(256) void conv_wmma(
    const unsigned short* __restrict__ fb, const unsigned short* __restrict__ wT,
    const int* __restrict__ nbr, float* __restrict__ xout) {
    __shared__ __align__(16) unsigned short sA[2][MT * C];  // 2 x 16 KB
    int n0   = blockIdx.x * MT;
    int t    = threadIdx.x;
    int lane = t & 31, wv = t >> 5;
    int lrow = t >> 4, lseg = t & 15;  // cooperative loader role
    int cl = lane & 15;
    bool hiL = lane >= 16;
    int c = wv * 16 + cl;

    auto issue_tap = [&](int tap, int buf) {
#pragma unroll
        for (int rr = 0; rr < 4; ++rr) {
            int row = lrow + rr * 16;
            int j = nbr[(size_t)(n0 + row) * 27 + tap];
            async_copy16(fb + (size_t)j * C + lseg * 8, &sA[buf][row * C + lseg * 8]);
        }
    };

    v8f acc0 = {}, acc1 = {}, acc2 = {}, acc3 = {};
    issue_tap(0, 0);
    for (int tap = 0; tap < 27; ++tap) {
        wait_asynccnt0();     // this tap's tile is in LDS (this wave's copies)
        __syncthreads();      // ...and everyone else's; also orders prior reads
        if (tap < 26) issue_tap(tap + 1, (tap + 1) & 1);
        const unsigned short* sbuf = sA[tap & 1];
#pragma unroll
        for (int kc = 0; kc < 4; ++kc) {
            BF16x16 b;
            const unsigned short* bp =
                wT + ((size_t)tap * C + c) * C + kc * 32 + (hiL ? 8 : 0);
            b.h[0] = *(const us8*)bp;
            b.h[1] = *(const us8*)(bp + 16);
            int abase = kc * 32 + (hiL ? 8 : 0);
            BF16x16 a;
            a.h[0] = *(const us8*)&sbuf[(0 * 16 + cl) * C + abase];
            a.h[1] = *(const us8*)&sbuf[(0 * 16 + cl) * C + abase + 16];
            acc0 = __builtin_amdgcn_wmma_f32_16x16x32_bf16(false, a.v, false, b.v, (short)0, acc0, false, false);
            a.h[0] = *(const us8*)&sbuf[(1 * 16 + cl) * C + abase];
            a.h[1] = *(const us8*)&sbuf[(1 * 16 + cl) * C + abase + 16];
            acc1 = __builtin_amdgcn_wmma_f32_16x16x32_bf16(false, a.v, false, b.v, (short)0, acc1, false, false);
            a.h[0] = *(const us8*)&sbuf[(2 * 16 + cl) * C + abase];
            a.h[1] = *(const us8*)&sbuf[(2 * 16 + cl) * C + abase + 16];
            acc2 = __builtin_amdgcn_wmma_f32_16x16x32_bf16(false, a.v, false, b.v, (short)0, acc2, false, false);
            a.h[0] = *(const us8*)&sbuf[(3 * 16 + cl) * C + abase];
            a.h[1] = *(const us8*)&sbuf[(3 * 16 + cl) * C + abase + 16];
            acc3 = __builtin_amdgcn_wmma_f32_16x16x32_bf16(false, a.v, false, b.v, (short)0, acc3, false, false);
        }
    }
    int mb = hiL ? 8 : 0;
    v8f* accs[4] = {&acc0, &acc1, &acc2, &acc3};
#pragma unroll
    for (int m = 0; m < 4; ++m) {
        v8f av = *accs[m];
#pragma unroll
        for (int r = 0; r < 8; ++r) {
            int gn = n0 + m * 16 + r + mb;
            if (gn < N_PTS) xout[(size_t)gn * C + c] = av[r];
        }
    }
}

// ---------------------------------------------------------------- generic X @ W^T WMMA
// out = (X(bf16, N_PAD rows) @ W^T(bf16,[cout][cin]) + bias) * scale, 64-row tiles.
// X buffers are padded to N_PAD rows -> branch-free loader; stores guarded.
__global__ __launch_bounds__(256) void gemm_xwt(
    const unsigned short* __restrict__ X, const unsigned short* __restrict__ W,
    const float* __restrict__ bias, float scale,
    float* __restrict__ outF, unsigned short* __restrict__ outB) {
    __shared__ __align__(16) unsigned short sA[MT * C];
    int n0 = blockIdx.x * MT;
    int t = threadIdx.x;
    int lane = t & 31, wv = t >> 5;
    {
        int lrow = t >> 4, lseg = t & 15;
#pragma unroll
        for (int rr = 0; rr < 4; ++rr) {
            int row = lrow + rr * 16;
            *(us8*)&sA[row * C + lseg * 8] =
                *(const us8*)(X + (size_t)(n0 + row) * C + lseg * 8);
        }
    }
    __syncthreads();
    int cl = lane & 15;
    bool hiL = lane >= 16;
    int c = wv * 16 + cl;
    v8f acc0 = {}, acc1 = {}, acc2 = {}, acc3 = {};
#pragma unroll
    for (int kc = 0; kc < 4; ++kc) {
        BF16x16 b;
        const unsigned short* bp = W + (size_t)c * C + kc * 32 + (hiL ? 8 : 0);
        b.h[0] = *(const us8*)bp;
        b.h[1] = *(const us8*)(bp + 16);
        int abase = kc * 32 + (hiL ? 8 : 0);
        BF16x16 a;
        a.h[0] = *(const us8*)&sA[(0 * 16 + cl) * C + abase];
        a.h[1] = *(const us8*)&sA[(0 * 16 + cl) * C + abase + 16];
        acc0 = __builtin_amdgcn_wmma_f32_16x16x32_bf16(false, a.v, false, b.v, (short)0, acc0, false, false);
        a.h[0] = *(const us8*)&sA[(1 * 16 + cl) * C + abase];
        a.h[1] = *(const us8*)&sA[(1 * 16 + cl) * C + abase + 16];
        acc1 = __builtin_amdgcn_wmma_f32_16x16x32_bf16(false, a.v, false, b.v, (short)0, acc1, false, false);
        a.h[0] = *(const us8*)&sA[(2 * 16 + cl) * C + abase];
        a.h[1] = *(const us8*)&sA[(2 * 16 + cl) * C + abase + 16];
        acc2 = __builtin_amdgcn_wmma_f32_16x16x32_bf16(false, a.v, false, b.v, (short)0, acc2, false, false);
        a.h[0] = *(const us8*)&sA[(3 * 16 + cl) * C + abase];
        a.h[1] = *(const us8*)&sA[(3 * 16 + cl) * C + abase + 16];
        acc3 = __builtin_amdgcn_wmma_f32_16x16x32_bf16(false, a.v, false, b.v, (short)0, acc3, false, false);
    }
    float bv = bias ? bias[c] : 0.0f;
    int mb = hiL ? 8 : 0;
    v8f* accs[4] = {&acc0, &acc1, &acc2, &acc3};
#pragma unroll
    for (int m = 0; m < 4; ++m) {
        v8f av = *accs[m];
#pragma unroll
        for (int r = 0; r < 8; ++r) {
            int gn = n0 + m * 16 + r + mb;
            if (gn >= N_PTS) continue;
            float v = (av[r] + bv) * scale;
            size_t idx = (size_t)gn * C + c;
            if (outF) outF[idx] = v;
            if (outB) outB[idx] = f2bf(v);
        }
    }
}

// ---------------------------------------------------------------- batchnorm
__global__ void bn_reduce(const float* __restrict__ x, float* __restrict__ stat) {
    __shared__ float ssum[256], ssq[256];
    int t = threadIdx.x, c = t & 127, half = t >> 7;
    float s = 0.f, q = 0.f;
    for (int n = blockIdx.x * 2 + half; n < N_PTS; n += gridDim.x * 2) {
        float v = x[(size_t)n * C + c];
        s += v; q += v * v;
    }
    ssum[t] = s; ssq[t] = q;
    __syncthreads();
    if (t < 128) {
        atomicAdd(&stat[c], ssum[t] + ssum[t + 128]);
        atomicAdd(&stat[C + c], ssq[t] + ssq[t + 128]);
    }
}
__global__ void bn_final(const float* __restrict__ stat, const float* __restrict__ gamma,
                         const float* __restrict__ beta, float* __restrict__ ss) {
    int c = threadIdx.x;
    if (c >= C) return;
    float mean = stat[c] / (float)N_PTS;
    float var  = stat[C + c] / (float)N_PTS - mean * mean;
    float sc = gamma[c] * rsqrtf(var + 1e-5f);
    ss[c] = sc;
    ss[C + c] = beta[c] - mean * sc;
}
__global__ void bn_apply(const float* __restrict__ x, const float* __restrict__ ss,
                         float* __restrict__ xf, unsigned short* __restrict__ xb) {
    long i = (long)blockIdx.x * 256 + threadIdx.x;
    if (i >= (long)N_PTS * C) return;
    int c = (int)(i & 127);
    float v = x[i] * ss[c] + ss[C + c];
    v = fmaxf(v, 0.0f);
    xf[i] = v;
    xb[i] = f2bf(v);
}

// ---------------------------------------------------------------- batch softmax over N
__global__ void sm_stats(const float* __restrict__ probs, float* __restrict__ stat) {
    int bk = blockIdx.x, b = bk >> 5, k = bk & 31;
    __shared__ float red[256];
    int t = threadIdx.x;
    float m = -3.4e38f;
    for (int n = t; n < N_PER; n += 256)
        m = fmaxf(m, probs[((size_t)b * N_PER + n) * KK + k]);
    red[t] = m; __syncthreads();
    for (int s = 128; s > 0; s >>= 1) { if (t < s) red[t] = fmaxf(red[t], red[t + s]); __syncthreads(); }
    m = red[0]; __syncthreads();
    float sum = 0.f;
    for (int n = t; n < N_PER; n += 256)
        sum += __expf(probs[((size_t)b * N_PER + n) * KK + k] - m);
    red[t] = sum; __syncthreads();
    for (int s = 128; s > 0; s >>= 1) { if (t < s) red[t] += red[t + s]; __syncthreads(); }
    if (t == 0) { stat[bk * 2] = m; stat[bk * 2 + 1] = red[0]; }
}

#define CTX_CHUNK 2048
__global__ void ctx_accum(const float* __restrict__ xbn, const float* __restrict__ probs,
                          const float* __restrict__ stat, float* __restrict__ ctx) {
    int b = blockIdx.x;
    int n0 = blockIdx.y * CTX_CHUNK;
    int n1 = min(n0 + CTX_CHUNK, N_PER);
    int t = threadIdx.x;
    int k = t >> 3, c0 = (t & 7) * 16;
    float m = stat[(b * KK + k) * 2];
    float accv[16];
#pragma unroll
    for (int j = 0; j < 16; ++j) accv[j] = 0.f;
    for (int n = n0; n < n1; ++n) {
        size_t gn = (size_t)b * N_PER + n;
        float e = __expf(probs[gn * KK + k] - m);
        const float* xr = xbn + gn * C + c0;
#pragma unroll
        for (int j = 0; j < 16; ++j) accv[j] += e * xr[j];
    }
    float* dst = ctx + ((size_t)b * KK + k) * C + c0;
#pragma unroll
    for (int j = 0; j < 16; ++j) atomicAdd(&dst[j], accv[j]);
}
__global__ void ctx_scale(float* __restrict__ ctx, const float* __restrict__ stat) {
    int i = blockIdx.x * 256 + threadIdx.x;
    if (i >= NB * KK * C) return;
    int bk = i >> 7;
    ctx[i] /= stat[bk * 2 + 1];
}

// kh / vh : (4*32,128) = ctx @ Wk^T + bk ; ctx @ Wv^T + bv  (scalar, tiny)
__global__ void khvh_gemm(const float* __restrict__ ctx, const float* __restrict__ inw,
                          const float* __restrict__ inb, float* __restrict__ kh,
                          float* __restrict__ vh) {
    int i = blockIdx.x * 256 + threadIdx.x;
    if (i >= 2 * NB * KK * C) return;
    int sel = i / (NB * KK * C);
    int r = i % (NB * KK * C);
    int bk = r >> 7, c = r & 127;
    const float* wrow = inw + (size_t)(C + sel * C + c) * C;
    const float* cx = ctx + (size_t)bk * C;
    float s = inb[C + sel * C + c];
    for (int j = 0; j < C; ++j) s += cx[j] * wrow[j];
    (sel ? vh : kh)[(size_t)bk * C + c] = s;
}

// ---------------------------------------------------------------- attention (per point,head)
__global__ void attention(const float* __restrict__ q, const float* __restrict__ kh,
                          const float* __restrict__ vh, unsigned short* __restrict__ ob) {
    long t = (long)blockIdx.x * 256 + threadIdx.x;
    if (t >= (long)N_PTS * HEADS) return;
    int n = (int)(t >> 3), h = (int)(t & 7);
    int b = n / N_PER;
    const float* qp = q + (size_t)n * C + h * 16;
    float qr[16];
#pragma unroll
    for (int d = 0; d < 16; ++d) qr[d] = qp[d];
    float sc[32];
    float mx = -3.4e38f;
#pragma unroll
    for (int k = 0; k < 32; ++k) {
        const float* kp = kh + ((size_t)b * KK + k) * C + h * 16;
        float d = 0.f;
#pragma unroll
        for (int j = 0; j < 16; ++j) d += qr[j] * kp[j];
        sc[k] = d;
        mx = fmaxf(mx, d);
    }
    float se = 0.f;
#pragma unroll
    for (int k = 0; k < 32; ++k) { sc[k] = __expf(sc[k] - mx); se += sc[k]; }
    float inv = 1.0f / se;
#pragma unroll
    for (int d = 0; d < 16; ++d) {
        float od = 0.f;
#pragma unroll
        for (int k = 0; k < 32; ++k)
            od += sc[k] * vh[((size_t)b * KK + k) * C + h * 16 + d];
        ob[(size_t)n * C + h * 16 + d] = f2bf(od * inv);
    }
}

// ---------------------------------------------------------------- launcher
extern "C" void kernel_launch(void* const* d_in, const int* in_sizes, int n_in,
                              void* d_out, int out_size, void* d_ws, size_t ws_size,
                              hipStream_t stream) {
    const float* feats   = (const float*)d_in[0];
    const float* probs   = (const float*)d_in[1];
    const float* conv_w  = (const float*)d_in[2];
    const float* gamma   = (const float*)d_in[3];
    const float* beta    = (const float*)d_in[4];
    const float* in_w    = (const float*)d_in[5];
    const float* in_b    = (const float*)d_in[6];
    const float* out_w   = (const float*)d_in[7];
    const float* out_b   = (const float*)d_in[8];
    const float* bott_w  = (const float*)d_in[9];
    const int*   coords  = (const int*)d_in[10];

    char* ws = (char*)d_ws;
    size_t off = 0;
    auto carve = [&](size_t bytes) -> void* {
        void* p = ws + off;
        off = (off + bytes + 255) & ~(size_t)255;
        return p;
    };
    int*            grid   = (int*)carve((size_t)NB * GRID3 * GRID3 * GRID3 * 4);
    int*            nbr    = (int*)carve((size_t)N_PAD * 27 * 4);
    unsigned short* fb     = (unsigned short*)carve((size_t)(N_PTS + 1) * C * 2);
    unsigned short* wT     = (unsigned short*)carve((size_t)27 * C * C * 2);
    unsigned short* wq_b   = (unsigned short*)carve((size_t)C * C * 2);
    unsigned short* wo_b   = (unsigned short*)carve((size_t)C * C * 2);
    unsigned short* wb_b   = (unsigned short*)carve((size_t)C * C * 2);
    float*          xconv  = (float*)carve((size_t)N_PTS * C * 4);  // reused as q
    float*          bnstat = (float*)carve(2 * C * 4);
    float*          bnss   = (float*)carve(2 * C * 4);
    float*          xbnF   = (float*)carve((size_t)N_PTS * C * 4);
    unsigned short* xbnB   = (unsigned short*)carve((size_t)N_PAD * C * 2);
    float*          smstat = (float*)carve((size_t)NB * KK * 2 * 4);
    float*          ctx    = (float*)carve((size_t)NB * KK * C * 4);
    float*          kh     = (float*)carve((size_t)NB * KK * C * 4);
    float*          vh     = (float*)carve((size_t)NB * KK * C * 4);
    unsigned short* o_b    = (unsigned short*)carve((size_t)N_PAD * C * 2);
    unsigned short* ao_b   = (unsigned short*)carve((size_t)N_PAD * C * 2);
    float*          q      = xconv;  // alias: conv output dead after BN apply

    const long grid_n = (long)NB * GRID3 * GRID3 * GRID3;

    // --- neighbor structure
    fill_i32<<<2048, 256, 0, stream>>>(grid, -1, grid_n);
    scatter_grid<<<(N_PTS + 255) / 256, 256, 0, stream>>>(coords, grid);
    build_nbr<<<(int)(((long)N_PAD * 27 + 255) / 256), 256, 0, stream>>>(coords, grid, nbr);

    // --- conversions / weight prep
    feats_to_bf16<<<(int)(((long)(N_PTS + 1) * C + 255) / 256), 256, 0, stream>>>(feats, fb);
    prep_convw<<<(27 * C * C + 255) / 256, 256, 0, stream>>>(conv_w, wT);
    prep_small<<<(3 * C * C + 255) / 256, 256, 0, stream>>>(in_w, out_w, bott_w, wq_b, wo_b, wb_b);

    // --- sparse conv (WMMA, async double-buffered gathers)
    conv_wmma<<<TILES, 256, 0, stream>>>(fb, wT, nbr, xconv);

    // --- batchnorm + relu
    fill_f32<<<1, 256, 0, stream>>>(bnstat, 0.0f, 2 * C);
    bn_reduce<<<512, 256, 0, stream>>>(xconv, bnstat);
    bn_final<<<1, 128, 0, stream>>>(bnstat, gamma, beta, bnss);
    bn_apply<<<(int)(((long)N_PTS * C + 255) / 256), 256, 0, stream>>>(xconv, bnss, xbnF, xbnB);

    // --- point-softmax + ctx einsum
    sm_stats<<<NB * KK, 256, 0, stream>>>(probs, smstat);
    fill_f32<<<64, 256, 0, stream>>>(ctx, 0.0f, NB * KK * C);
    dim3 cgrid(NB, (N_PER + CTX_CHUNK - 1) / CTX_CHUNK);
    ctx_accum<<<cgrid, 256, 0, stream>>>(xbnF, probs, smstat, ctx);
    ctx_scale<<<(NB * KK * C + 255) / 256, 256, 0, stream>>>(ctx, smstat);

    // --- kh/vh (tiny), q (WMMA), attention
    khvh_gemm<<<(2 * NB * KK * C + 255) / 256, 256, 0, stream>>>(ctx, in_w, in_b, kh, vh);
    gemm_xwt<<<TILES, 256, 0, stream>>>(xbnB, wq_b, in_b, 0.25f, q, (unsigned short*)nullptr);
    attention<<<(int)(((long)N_PTS * HEADS + 255) / 256), 256, 0, stream>>>(q, kh, vh, o_b);

    // --- out proj + bottleneck (WMMA)
    gemm_xwt<<<TILES, 256, 0, stream>>>(o_b, wo_b, out_b, 1.0f, (float*)nullptr, ao_b);
    gemm_xwt<<<TILES, 256, 0, stream>>>(ao_b, wb_b, (const float*)nullptr, 1.0f,
                                        (float*)d_out, (unsigned short*)nullptr);
}
